// PointSIS_Seg_Model_66099546685663
// MI455X (gfx1250) — matrix operand order repaired and checked by reading, not compiled, not run on hardware
//
#include <hip/hip_runtime.h>
#include <hip/hip_bf16.h>

// ---------------- model dimensions (from reference) ----------------
#define BB     4
#define GG     256
#define OS_    2
#define LVL_   3
#define DD     128
#define DI_    256
#define DS_    16
#define DTR_   8
#define DCONV_ 4
#define GS_    64
#define NQ_    100
#define NCLS_  21
#define HH     8
#define NP_    (GG*GS_)   // 16384
#define BG_    (BB*GG)    // 1024
#define TOKMAX (BB*LVL_*GG) // 3072

typedef __attribute__((ext_vector_type(16))) _Float16 v16h;
typedef __attribute__((ext_vector_type(8)))  _Float16 v8h;
typedef __attribute__((ext_vector_type(8)))  float    v8f;

// ---------------- device helpers ----------------
__device__ __forceinline__ float gelu_f(float x){
  float x3 = x*x*x;
  return 0.5f*x*(1.f + tanhf(0.7978845608028654f*(x + 0.044715f*x3)));
}
__device__ __forceinline__ float silu_f(float x){ return x/(1.f + __expf(-x)); }
__device__ __forceinline__ float softplus_f(float x){ return (x > 20.f) ? x : log1pf(__expf(x)); }

template<int ACT> __device__ __forceinline__ float act_f(float x){
  if (ACT == 1) return gelu_f(x);
  if (ACT == 2) return silu_f(x);
  if (ACT == 3) return softplus_f(x);
  return x;
}

// ---------------- WMMA GEMM: C[M,N] = act(A[M,K] * W + bias) ----------------
// Block = 256 threads (8 wave32), block tile 128x16, K-step 32.
// Fast path: float4 global loads -> f16 LDS. Edge path: batched clamped loads
// (load phase into registers, single wait, then convert/mask/store) to avoid
// per-element load-wait serialization. Fragments read back as contiguous
// ds_load_b128 pairs per CDNA5 ISA 7.12.2; one v_wmma per wave per K-step.
#define BM_ 128
#define BN_ 16
#define BK_ 32

template<int ACT, bool TRANSB>
__global__ __launch_bounds__(256)
void gemm_wmma_kernel(const float* __restrict__ A, const float* __restrict__ W,
                      const float* __restrict__ bias, float* __restrict__ C,
                      int M, int N, int K, int lda, int ldc)
{
  __shared__ __align__(16) _Float16 As[BM_][BK_];   // row-major (M x K)
  __shared__ __align__(16) _Float16 Bs[BN_][BK_];   // N-major  (N x K)

  const int tiles_n = (N + BN_ - 1) / BN_;
  const int tm = (int)blockIdx.x / tiles_n;
  const int tn = (int)blockIdx.x % tiles_n;
  const int t  = threadIdx.x;
  const int w    = t >> 5;        // wave id 0..7 -> M sub-tile
  const int lane = t & 31;
  const int hi   = lane >> 4;     // 0: lanes 0-15, 1: lanes 16-31
  const int l15  = lane & 15;

  const int m0 = tm * BM_;
  const int n0 = tn * BN_;
  const bool mfull = (m0 + BM_) <= M;
  const bool nfull = (n0 + BN_) <= N;
  const bool lda4  = (lda & 3) == 0;
  const bool kvec  = TRANSB ? ((K & 3) == 0) : ((N & 3) == 0);

  v8f acc = {};
  for (int kk = 0; kk < K; kk += BK_) {
    const bool kfull = (kk + BK_) <= K;

    // ---- stage A tile: 128x32 floats, 4 float4-chunks per thread ----
    if (mfull && kfull && lda4) {
      const float* ap = A + (size_t)m0*lda + kk;
      if (kk + 2*BK_ <= K)  // one prefetch per thread for the next K slab
        __builtin_prefetch(ap + (size_t)(t >> 1)*lda + ((t & 1) << 4) + BK_, 0, 0);
#pragma unroll
      for (int c = 0; c < 4; ++c) {
        int id  = c*256 + t;            // 0..1023
        int row = id >> 3;              // 0..127
        int c4  = (id & 7) << 2;        // 0,4,...,28
        const float4 f = *(const float4*)(ap + (size_t)row*lda + c4);
        As[row][c4+0] = (_Float16)f.x;  As[row][c4+1] = (_Float16)f.y;
        As[row][c4+2] = (_Float16)f.z;  As[row][c4+3] = (_Float16)f.w;
      }
    } else {
      // batched clamped loads: issue all 16 loads, then convert/mask/store
      float tmp[16];
#pragma unroll
      for (int c = 0; c < 4; ++c) {
        int id  = c*256 + t;
        int row = id >> 3;
        int c4  = (id & 7) << 2;
        int grow  = m0 + row;
        int growc = (grow < M) ? grow : 0;
        const float* base = A + (size_t)growc*lda;
#pragma unroll
        for (int j = 0; j < 4; ++j) {
          int k = kk + c4 + j;
          tmp[c*4+j] = base[(k < K) ? k : 0];
        }
      }
#pragma unroll
      for (int c = 0; c < 4; ++c) {
        int id  = c*256 + t;
        int row = id >> 3;
        int c4  = (id & 7) << 2;
        bool rok = (m0 + row) < M;
#pragma unroll
        for (int j = 0; j < 4; ++j) {
          int k = kk + c4 + j;
          As[row][c4+j] = (_Float16)((rok && k < K) ? tmp[c*4+j] : 0.f);
        }
      }
    }

    // ---- stage B tile: 16(N) x 32(K) floats, threads 0..127, one float4 each ----
    if (t < 128) {
      if (TRANSB) {
        // W is [N][K] row-major; contiguous along K
        int n  = t >> 3;                // 0..15
        int k4 = (t & 7) << 2;          // 0..28
        if (nfull && kfull && kvec) {
          const float4 f = *(const float4*)(W + (size_t)(n0+n)*K + kk + k4);
          Bs[n][k4+0] = (_Float16)f.x;  Bs[n][k4+1] = (_Float16)f.y;
          Bs[n][k4+2] = (_Float16)f.z;  Bs[n][k4+3] = (_Float16)f.w;
        } else {
          int gn  = n0 + n;
          int gnc = (gn < N) ? gn : 0;
          const float* base = W + (size_t)gnc*K;
          float tb[4];
#pragma unroll
          for (int j = 0; j < 4; ++j) {
            int k = kk + k4 + j;
            tb[j] = base[(k < K) ? k : 0];
          }
          bool nok = gn < N;
#pragma unroll
          for (int j = 0; j < 4; ++j) {
            int k = kk + k4 + j;
            Bs[n][k4+j] = (_Float16)((nok && k < K) ? tb[j] : 0.f);
          }
        }
      } else {
        // W is [K][N] row-major; load along N, store transposed
        int krow = t >> 2;              // 0..31
        int nc4  = (t & 3) << 2;        // 0,4,8,12
        if (nfull && kfull && kvec) {
          const float4 f = *(const float4*)(W + (size_t)(kk+krow)*N + n0 + nc4);
          Bs[nc4+0][krow] = (_Float16)f.x;  Bs[nc4+1][krow] = (_Float16)f.y;
          Bs[nc4+2][krow] = (_Float16)f.z;  Bs[nc4+3][krow] = (_Float16)f.w;
        } else {
          int k  = kk + krow;
          int kc = (k < K) ? k : 0;
          const float* base = W + (size_t)kc*N;
          float tb[4];
#pragma unroll
          for (int j = 0; j < 4; ++j) {
            int gn = n0 + nc4 + j;
            tb[j] = base[(gn < N) ? gn : 0];
          }
          bool kok = k < K;
#pragma unroll
          for (int j = 0; j < 4; ++j) {
            int gn = n0 + nc4 + j;
            Bs[nc4+j][krow] = (_Float16)((kok && gn < N) ? tb[j] : 0.f);
          }
        }
      }
    }
    __syncthreads();

    // ---- fragments from LDS (two contiguous b128 runs each, ISA 7.12.2) ----
    const int arow = w*16 + l15;
    v8h alo = *(const v8h*)&As[arow][hi*8];
    v8h ahi = *(const v8h*)&As[arow][16 + hi*8];
    v16h af = __builtin_shufflevector(alo, ahi, 0,1,2,3,4,5,6,7,8,9,10,11,12,13,14,15);
    v8h blo = *(const v8h*)&Bs[l15][hi*8];
    v8h bhi = *(const v8h*)&Bs[l15][16 + hi*8];
    v16h bf = __builtin_shufflevector(blo, bhi, 0,1,2,3,4,5,6,7,8,9,10,11,12,13,14,15);

    acc = __builtin_amdgcn_wmma_f32_16x16x32_f16(false, af, false, bf,
                                                 (short)0, acc, false, false);
    __syncthreads();
  }

  // ---- store: C/D layout vgpr r -> row = 8*hi + r, col = l15 ----
  const int ocol = n0 + l15;
#pragma unroll
  for (int r = 0; r < 8; ++r) {
    int orow = m0 + w*16 + hi*8 + r;
    if (orow < M && ocol < N) {
      float v = acc[r];
      if (bias) v += bias[ocol];
      C[(size_t)orow*ldc + ocol] = act_f<ACT>(v);
    }
  }
}

// ---------------- LayerNorm over D=128 (optional residual add, optional GELU) ----------------
template<bool GELU>
__global__ __launch_bounds__(128)
void ln_kernel(const float* __restrict__ x, const float* __restrict__ res,
               const float* __restrict__ g, const float* __restrict__ b,
               float* __restrict__ out)
{
  __shared__ float s1[128], s2[128];
  int r = blockIdx.x, t = threadIdx.x;
  float v = x[(size_t)r*DD + t];
  if (res) v += res[(size_t)r*DD + t];
  s1[t] = v; s2[t] = v*v;
  __syncthreads();
  for (int o = 64; o > 0; o >>= 1) {
    if (t < o) { s1[t] += s1[t+o]; s2[t] += s2[t+o]; }
    __syncthreads();
  }
  float mean = s1[0] * (1.f/128.f);
  float var  = s2[0] * (1.f/128.f) - mean*mean;
  float y = (v - mean) * rsqrtf(var + 1e-5f) * g[t] + b[t];
  if (GELU) y = gelu_f(y);
  out[(size_t)r*DD + t] = y;
}

// ---------------- elementwise ----------------
__global__ void add_kernel(const float* __restrict__ a, const float* __restrict__ b,
                           float* __restrict__ c){
  size_t i = (size_t)blockIdx.x*blockDim.x + threadIdx.x;
  c[i] = a[i] + b[i];
}
__global__ void concat2_kernel(const float* __restrict__ a, const float* __restrict__ b,
                               float* __restrict__ c){
  int r = blockIdx.x, t = threadIdx.x;
  c[(size_t)r*256 + t]       = a[(size_t)r*128 + t];
  c[(size_t)r*256 + 128 + t] = b[(size_t)r*128 + t];
}

// ---------------- Hilbert-order scatter/gather ----------------
__global__ void scatter_ho_kernel(const float* __restrict__ h, const int* __restrict__ orders,
                                  float* __restrict__ ho){
  int o = blockIdx.x / BG_, i = blockIdx.x % BG_;
  int j = orders[o*BG_ + i];
  int b = j >> 8, g = j & 255;
  ho[((size_t)(b*(OS_*GG) + o*GG + g))*128 + threadIdx.x] = h[(size_t)i*128 + threadIdx.x];
}
__global__ void gather_ho_kernel(const float* __restrict__ mixed, const int* __restrict__ orders,
                                 float* __restrict__ hog){
  int o = blockIdx.x / BG_, j = blockIdx.x % BG_;
  int i = orders[o*BG_ + j];
  int b = i >> 8, g = i & 255;
  hog[(size_t)j*(OS_*128) + o*128 + threadIdx.x] =
      mixed[((size_t)(b*(OS_*GG) + o*GG + g))*128 + threadIdx.x];
}
__global__ void scatter_enc_kernel(const float* __restrict__ f, const int* __restrict__ orders_o,
                                   float* __restrict__ cat, int lvl){
  int i = blockIdx.x;
  int j = orders_o[i];
  int b = j >> 8, g = j & 255;
  cat[((size_t)(b*(LVL_*GG) + lvl*GG + g))*128 + threadIdx.x] = f[(size_t)i*128 + threadIdx.x];
}
__global__ void gather_enc_kernel(const float* __restrict__ cat, const int* __restrict__ orders_o,
                                  float* __restrict__ f, int lvl){
  int j = blockIdx.x;
  int i = orders_o[j];
  int b = i >> 8, g = i & 255;
  f[(size_t)j*128 + threadIdx.x] = cat[((size_t)(b*(LVL_*GG) + lvl*GG + g))*128 + threadIdx.x];
}

// ---------------- Mamba: causal depthwise conv (DCONV=4) + SiLU ----------------
__global__ void conv_silu_kernel(const float* __restrict__ xz, const float* __restrict__ cw,
                                 const float* __restrict__ cb, float* __restrict__ out,
                                 int Bm, int L){
  int idx = blockIdx.x*blockDim.x + threadIdx.x;
  int total = Bm*L*DI_;
  if (idx >= total) return;
  int d = idx % DI_;
  int l = (idx / DI_) % L;
  int b = idx / (DI_*L);
  float acc = cb[d];
#pragma unroll
  for (int j = 0; j < DCONV_; ++j) {
    int li = l - (DCONV_-1) + j;
    if (li >= 0) acc += xz[((size_t)(b*L + li))*512 + d] * cw[d*DCONV_ + j];
  }
  out[idx] = silu_f(acc);
}

// ---------------- Mamba selective scan (fused D-skip + z-gating) ----------------
__global__ void scan_kernel(const float* __restrict__ dt, const float* __restrict__ dbc,
                            const float* __restrict__ xi, const float* __restrict__ xz,
                            const float* __restrict__ A_log, const float* __restrict__ Dp,
                            float* __restrict__ yg, int Bm, int L){
  int id = blockIdx.x*blockDim.x + threadIdx.x;
  if (id >= Bm*DI_) return;
  int d = id % DI_, b = id / DI_;
  float A[DS_], h[DS_];
#pragma unroll
  for (int s = 0; s < DS_; ++s) { A[s] = -__expf(A_log[d*DS_ + s]); h[s] = 0.f; }
  float Dd = Dp[d];
  for (int t = 0; t < L; ++t) {
    size_t tok = (size_t)b*L + t;
    float dtv = dt[tok*DI_ + d];
    float xv  = xi[tok*DI_ + d];
    float y = 0.f;
#pragma unroll
    for (int s = 0; s < DS_; ++s) {
      float dA = __expf(dtv*A[s]);
      float Bv = dbc[tok*48 + DTR_ + s];
      float Cv = dbc[tok*48 + DTR_ + DS_ + s];
      h[s] = dA*h[s] + dtv*Bv*xv;
      y += h[s]*Cv;
    }
    y += xv*Dd;
    float z = xz[tok*512 + DI_ + d];
    yg[tok*DI_ + d] = y * silu_f(z);
  }
}

// ---------------- decoder helpers ----------------
__global__ void q_init_kernel(const float* __restrict__ qe, float* __restrict__ q){
  int r = blockIdx.x;
  q[(size_t)r*DD + threadIdx.x] = qe[(size_t)(r % NQ_)*DD + threadIdx.x];
}
__global__ void add_qpos_kernel(const float* __restrict__ q, const float* __restrict__ qp,
                                float* __restrict__ out){
  int r = blockIdx.x;
  out[(size_t)r*DD + threadIdx.x] =
      q[(size_t)r*DD + threadIdx.x] + qp[(size_t)(r % NQ_)*DD + threadIdx.x];
}

__global__ __launch_bounds__(256)
void mha_kernel(const float* __restrict__ Q, const float* __restrict__ Km,
                const float* __restrict__ V, float* __restrict__ O){
  int bh = blockIdx.x;
  int b = bh >> 3, h = bh & 7;
  int t = threadIdx.x;  // key index g
  __shared__ float p[256], red[256], qv[16];
  __shared__ float Vs[256*16];
  float kvv[16];
#pragma unroll
  for (int d = 0; d < 16; ++d) {
    kvv[d]       = Km[((size_t)(b*GG + t))*DD + h*16 + d];
    Vs[t*16 + d] = V [((size_t)(b*GG + t))*DD + h*16 + d];
  }
  __syncthreads();
  const float scale = 0.25f; // 1/sqrt(16)
  for (int qi = 0; qi < NQ_; ++qi) {
    if (t < 16) qv[t] = Q[((size_t)(b*NQ_ + qi))*DD + h*16 + t];
    __syncthreads();
    float s = 0.f;
#pragma unroll
    for (int d = 0; d < 16; ++d) s += qv[d]*kvv[d];
    s *= scale;
    red[t] = s; __syncthreads();
    for (int o = 128; o > 0; o >>= 1) { if (t < o) red[t] = fmaxf(red[t], red[t+o]); __syncthreads(); }
    float m = red[0]; __syncthreads();
    float e = __expf(s - m);
    red[t] = e; __syncthreads();
    for (int o = 128; o > 0; o >>= 1) { if (t < o) red[t] += red[t+o]; __syncthreads(); }
    float sum = red[0];
    p[t] = e / sum;
    __syncthreads();
    if (t < 16) {
      float acc = 0.f;
      for (int g = 0; g < GG; ++g) acc += p[g]*Vs[g*16 + t];
      O[((size_t)(b*NQ_ + qi))*DD + h*16 + t] = acc;
    }
    __syncthreads();
  }
}

// ---------------- top-64 IDW interpolation: out[b*NP+n, q] ----------------
__global__ __launch_bounds__(256)
void interp_kernel(const float* __restrict__ pc, const float* __restrict__ sc,
                   const float* __restrict__ pm, float* __restrict__ out){
  int blk = blockIdx.x;
  int b = blk / NP_, n = blk % NP_;
  int t = threadIdx.x;  // source point g
  __shared__ float d2s[256], w[256], red[256];
  float nx = pc[((size_t)(b*NP_ + n))*3 + 0];
  float ny = pc[((size_t)(b*NP_ + n))*3 + 1];
  float nz = pc[((size_t)(b*NP_ + n))*3 + 2];
  float dx = nx - sc[((size_t)(b*GG + t))*3 + 0];
  float dy = ny - sc[((size_t)(b*GG + t))*3 + 1];
  float dz = nz - sc[((size_t)(b*GG + t))*3 + 2];
  float d2 = dx*dx + dy*dy + dz*dz;
  d2s[t] = d2;
  __syncthreads();
  int rank = 0;
  for (int g = 0; g < GG; ++g) {
    float o = d2s[g];
    rank += (o < d2) || (o == d2 && g < t);
  }
  float wt = (rank < GS_) ? 1.f/(sqrtf(fmaxf(d2, 1e-12f)) + 1e-8f) : 0.f;
  w[t] = wt; red[t] = wt;
  __syncthreads();
  for (int o = 128; o > 0; o >>= 1) { if (t < o) red[t] += red[t+o]; __syncthreads(); }
  float wsum = red[0];
  if (t < NQ_) {
    float acc = 0.f;
    for (int g = 0; g < GG; ++g) acc += w[g]*pm[((size_t)(b*NQ_ + t))*GG + g];
    out[((size_t)(b*NP_ + n))*NQ_ + t] = acc / wsum;
  }
}

// ================== host-side orchestration ==================
struct BlockP {
  const float *A_log, *Dp, *conv_b, *conv_w, *dt_b, *dt_w, *in_proj,
              *norm_b, *norm_g, *out_proj, *x_proj;
};
struct MixerP { BlockP blk[2]; const float *nf_b, *nf_g; };
struct FuseP  { const float *l1b, *l1w, *l2b, *l2w, *lnb, *lng; };
struct DecL   { const float *f0b,*f0w,*f1b,*f1w,*ln1b,*ln1g,*ln2b,*ln2g,*wk,*wo,*wq,*wv; };

struct Scratch {
  float *t256, *pos, *h0, *hp, *t1, *t2, *hs[3], *hog;
  float *seq_in, *seq_out, *lnt, *x1, *r2, *x2;
  float *xz, *conv, *dbc, *dt, *yg;
  float *q, *qp, *q2, *Qp, *Kp, *Vp, *attn, *ao, *t512, *ff, *t128, *memb, *pm;
};

static inline void gemm(hipStream_t st, const float* A, const float* W, const float* bias,
                        float* C, int M, int N, int K, int lda, int ldc,
                        int act = 0, bool transb = false)
{
  dim3 grid((unsigned)(((M + BM_ - 1)/BM_) * ((N + BN_ - 1)/BN_)));
  dim3 blk(256);
  if (transb) { gemm_wmma_kernel<0,true ><<<grid,blk,0,st>>>(A,W,bias,C,M,N,K,lda,ldc); return; }
  switch (act) {
    case 1:  gemm_wmma_kernel<1,false><<<grid,blk,0,st>>>(A,W,bias,C,M,N,K,lda,ldc); break;
    case 2:  gemm_wmma_kernel<2,false><<<grid,blk,0,st>>>(A,W,bias,C,M,N,K,lda,ldc); break;
    case 3:  gemm_wmma_kernel<3,false><<<grid,blk,0,st>>>(A,W,bias,C,M,N,K,lda,ldc); break;
    default: gemm_wmma_kernel<0,false><<<grid,blk,0,st>>>(A,W,bias,C,M,N,K,lda,ldc); break;
  }
}
static inline void ln(hipStream_t st, const float* x, const float* res,
                      const float* g, const float* b, float* out, int rows, bool gelu)
{
  if (gelu) ln_kernel<true ><<<rows,128,0,st>>>(x,res,g,b,out);
  else      ln_kernel<false><<<rows,128,0,st>>>(x,res,g,b,out);
}

static void run_mamba(hipStream_t st, const BlockP& p, const float* in, float* out,
                      int Bm, int L, Scratch& S)
{
  int T = Bm*L;
  gemm(st, in,     p.in_proj,  nullptr, S.xz,  T, 2*DI_, DD,  DD,  2*DI_, 0); // in_proj
  conv_silu_kernel<<<(T*DI_ + 255)/256, 256, 0, st>>>(S.xz, p.conv_w, p.conv_b, S.conv, Bm, L);
  gemm(st, S.conv, p.x_proj,   nullptr, S.dbc, T, 48,    DI_, DI_, 48,    0); // x_proj
  gemm(st, S.dbc,  p.dt_w,     p.dt_b,  S.dt,  T, DI_,   DTR_,48,  DI_,   3); // softplus(dt@W+b)
  scan_kernel<<<(Bm*DI_ + 255)/256, 256, 0, st>>>(S.dt, S.dbc, S.conv, S.xz,
                                                  p.A_log, p.Dp, S.yg, Bm, L);
  gemm(st, S.yg,   p.out_proj, nullptr, out,   T, DD,    DI_, DI_, DD,    0); // out_proj
}

static void run_mixer(hipStream_t st, const MixerP& m, const float* x, float* out,
                      int Bm, int L, Scratch& S)
{
  int T = Bm*L;
  ln(st, x, nullptr, m.blk[0].norm_g, m.blk[0].norm_b, S.lnt, T, false);
  run_mamba(st, m.blk[0], S.lnt, S.x1, Bm, L, S);
  add_kernel<<<T, 128, 0, st>>>(S.x1, x, S.r2);                 // r2 = mamba0 + x
  ln(st, S.r2, nullptr, m.blk[1].norm_g, m.blk[1].norm_b, S.lnt, T, false);
  run_mamba(st, m.blk[1], S.lnt, S.x2, Bm, L, S);
  ln(st, S.x2, S.r2, m.nf_g, m.nf_b, out, T, false);            // LN_f(x2 + r2)
}

extern "C" void kernel_launch(void* const* d_in, const int* in_sizes, int n_in,
                              void* d_out, int out_size, void* d_ws, size_t ws_size,
                              hipStream_t stream)
{
  (void)in_sizes; (void)n_in; (void)out_size; (void)ws_size;
  const float* s_coord = (const float*)d_in[0];
  const float* s_feat  = (const float*)d_in[1];
  const float* parent  = (const float*)d_in[2];
  const int*   orders  = (const int*)  d_in[3];

  // -------- params: jax pytree flatten order (dicts sorted by key) --------
  int ci = 4;
  auto nf = [&]() -> const float* { return (const float*)d_in[ci++]; };

  const float* cls_b = nf(); const float* cls_w = nf();                 // 'cls'
  DecL dec[LVL_-1];                                                     // 'dec_layers'
  for (int l = 0; l < LVL_-1; ++l) {
    dec[l].f0b = nf(); dec[l].f0w = nf(); dec[l].f1b = nf(); dec[l].f1w = nf();
    dec[l].ln1b = nf(); dec[l].ln1g = nf(); dec[l].ln2b = nf(); dec[l].ln2g = nf();
    dec[l].wk = nf(); dec[l].wo = nf(); dec[l].wq = nf(); dec[l].wv = nf();
  }
  auto parse_block = [&](BlockP& b){
    b.A_log = nf(); b.Dp = nf(); b.conv_b = nf(); b.conv_w = nf();
    b.dt_b = nf(); b.dt_w = nf(); b.in_proj = nf();
    b.norm_b = nf(); b.norm_g = nf(); b.out_proj = nf(); b.x_proj = nf();
  };
  auto parse_mixer = [&](MixerP& m){
    parse_block(m.blk[0]); parse_block(m.blk[1]); m.nf_b = nf(); m.nf_g = nf();
  };
  MixerP encm[OS_];  for (int o = 0; o < OS_; ++o) parse_mixer(encm[o]); // 'enc_mixers'
  const float *fe0_b = nf(), *fe0_w = nf(), *fe1_b = nf(), *fe1_w = nf(); // 'feat_enc'
  FuseP fe; fe.l1b = nf(); fe.l1w = nf(); fe.l2b = nf(); fe.l2w = nf(); fe.lnb = nf(); fe.lng = nf(); // 'fuse_e'
  FuseP fo; fo.l1b = nf(); fo.l1w = nf(); fo.l2b = nf(); fo.l2w = nf(); fo.lnb = nf(); fo.lng = nf(); // 'fuse_o'
  MixerP fxm[LVL_];  for (int l = 0; l < LVL_; ++l) parse_mixer(fxm[l]); // 'fx_mixers'
  const float *mh0_b = nf(), *mh0_w = nf(), *mh1_b = nf(), *mh1_w = nf(); // 'mask_head'
  const float *pe0_b = nf(), *pe0_w = nf(), *pe1_b = nf(), *pe1_w = nf(); // 'pos_enc'
  const float* qe   = nf();                                               // 'query_embed'
  const float* qpos = nf();                                               // 'query_pos'

  // -------- workspace carve --------
  char* wp = (char*)d_ws;
  auto alloc = [&](size_t nfl) -> float* {
    float* p = (float*)wp; wp += ((nfl*sizeof(float) + 255)/256)*256; return p;
  };
  Scratch S;
  S.t256 = alloc((size_t)BG_*256);  S.pos = alloc((size_t)BG_*DD);  S.h0 = alloc((size_t)BG_*DD);
  S.hp = alloc((size_t)BG_*256);    S.t1 = alloc((size_t)BG_*DD);   S.t2 = alloc((size_t)BG_*DD);
  for (int i = 0; i < 3; ++i) S.hs[i] = alloc((size_t)BG_*DD);
  S.hog = alloc((size_t)BG_*OS_*DD);
  S.seq_in  = alloc((size_t)TOKMAX*DD);  S.seq_out = alloc((size_t)TOKMAX*DD);
  S.lnt = alloc((size_t)TOKMAX*DD); S.x1 = alloc((size_t)TOKMAX*DD);
  S.r2  = alloc((size_t)TOKMAX*DD); S.x2 = alloc((size_t)TOKMAX*DD);
  S.xz  = alloc((size_t)TOKMAX*2*DI_); S.conv = alloc((size_t)TOKMAX*DI_);
  S.dbc = alloc((size_t)TOKMAX*48);    S.dt   = alloc((size_t)TOKMAX*DI_);
  S.yg  = alloc((size_t)TOKMAX*DI_);
  S.q = alloc((size_t)BB*NQ_*DD); S.qp = alloc((size_t)BB*NQ_*DD); S.q2 = alloc((size_t)BB*NQ_*DD);
  S.Qp = alloc((size_t)BB*NQ_*DD); S.Kp = alloc((size_t)BG_*DD);   S.Vp = alloc((size_t)BG_*DD);
  S.attn = alloc((size_t)BB*NQ_*DD); S.ao = alloc((size_t)BB*NQ_*DD);
  S.t512 = alloc((size_t)BB*NQ_*4*DD); S.ff = alloc((size_t)BB*NQ_*DD);
  S.t128 = alloc((size_t)BB*NQ_*DD);   S.memb = alloc((size_t)BB*NQ_*DD);
  S.pm   = alloc((size_t)BB*NQ_*GG);

  float* out1 = (float*)d_out;                       // (B*NP, NQ)
  float* out2 = out1 + (size_t)BB*NP_*NQ_;           // (B, NQ, NCLS)
  hipStream_t st = stream;

  // -------- point encoders --------
  gemm(st, s_coord, pe0_w, pe0_b, S.t256, BG_, 256, 3,   3,   256, 1);
  gemm(st, S.t256,  pe1_w, pe1_b, S.pos,  BG_, DD,  256, 256, DD,  0);
  gemm(st, s_feat,  fe0_w, fe0_b, S.t256, BG_, 256, 4,   4,   256, 1);
  gemm(st, S.t256,  fe1_w, fe1_b, S.h0,   BG_, DD,  256, 256, DD,  0);
  float* h = S.h0;

  // -------- fx levels --------
  for (int lvl = 0; lvl < LVL_; ++lvl) {
    concat2_kernel<<<BG_, 128, 0, st>>>(h, S.pos, S.hp);
    gemm(st, S.hp, fe.l1w, fe.l1b, S.t1, BG_, DD, 2*DD, 2*DD, DD, 0);
    ln(st, S.t1, nullptr, fe.lng, fe.lnb, S.t2, BG_, true);   // gelu(LN(x))
    gemm(st, S.t2, fe.l2w, fe.l2b, h, BG_, DD, DD, DD, DD, 0);

    scatter_ho_kernel<<<OS_*BG_, 128, 0, st>>>(h, orders, S.seq_in);
    run_mixer(st, fxm[lvl], S.seq_in, S.seq_out, BB, OS_*GG, S);
    gather_ho_kernel<<<OS_*BG_, 128, 0, st>>>(S.seq_out, orders, S.hog);

    gemm(st, S.hog, fo.l1w, fo.l1b, S.t1, BG_, DD, OS_*DD, OS_*DD, DD, 0);
    ln(st, S.t1, nullptr, fo.lng, fo.lnb, S.t2, BG_, true);
    gemm(st, S.t2, fo.l2w, fo.l2b, S.hs[lvl], BG_, DD, DD, DD, DD, 0);
    h = S.hs[lvl];
  }

  // -------- cross-level encoder refinement --------
  float* lf[LVL_] = { S.hs[2], S.hs[1], S.hs[0] };
  for (int o = 0; o < OS_; ++o) {
    for (int lvl = 0; lvl < LVL_; ++lvl)
      scatter_enc_kernel<<<BG_, 128, 0, st>>>(lf[lvl], orders + o*BG_, S.seq_in, lvl);
    run_mixer(st, encm[o], S.seq_in, S.seq_out, BB, LVL_*GG, S);
    for (int lvl = 0; lvl < LVL_; ++lvl)
      gather_enc_kernel<<<BG_, 128, 0, st>>>(S.seq_out, orders + o*BG_, lf[lvl], lvl);
  }

  // -------- transformer decoder (2 layers) --------
  q_init_kernel<<<BB*NQ_, 128, 0, st>>>(qe, S.q);
  float* qcur = S.q;
  for (int li = 0; li < LVL_-1; ++li) {
    const DecL& dl = dec[li];
    const float* kv = lf[li];
    add_qpos_kernel<<<BB*NQ_, 128, 0, st>>>(qcur, qpos, S.qp);
    gemm(st, S.qp, dl.wq, nullptr, S.Qp, BB*NQ_, DD, DD, DD, DD, 0);
    gemm(st, kv,   dl.wk, nullptr, S.Kp, BG_,    DD, DD, DD, DD, 0);
    gemm(st, kv,   dl.wv, nullptr, S.Vp, BG_,    DD, DD, DD, DD, 0);
    mha_kernel<<<BB*HH, 256, 0, st>>>(S.Qp, S.Kp, S.Vp, S.attn);
    gemm(st, S.attn, dl.wo, nullptr, S.ao, BB*NQ_, DD, DD, DD, DD, 0);
    ln(st, S.ao, qcur, dl.ln1g, dl.ln1b, S.q2, BB*NQ_, false);       // LN1(q + attn)
    gemm(st, S.q2,   dl.f0w, dl.f0b, S.t512, BB*NQ_, 4*DD, DD,  DD,   4*DD, 1);
    gemm(st, S.t512, dl.f1w, dl.f1b, S.ff,   BB*NQ_, DD,   4*DD,4*DD, DD,   0);
    ln(st, S.ff, S.q2, dl.ln2g, dl.ln2b, S.q, BB*NQ_, false);        // LN2(q + ffn)
    qcur = S.q;
  }

  // -------- mask head + pred_mask + class logits --------
  gemm(st, qcur,   mh0_w, mh0_b, S.t128, BB*NQ_, DD, DD, DD, DD, 1);
  gemm(st, S.t128, mh1_w, mh1_b, S.memb, BB*NQ_, DD, DD, DD, DD, 0);
  for (int b = 0; b < BB; ++b)  // pred_mask[b] = mask_emb[b] @ point_emb[b]^T
    gemm(st, S.memb + (size_t)b*NQ_*DD, lf[LVL_-1] + (size_t)b*GG*DD, nullptr,
         S.pm + (size_t)b*NQ_*GG, NQ_, GG, DD, DD, GG, 0, /*transb=*/true);
  gemm(st, qcur, cls_w, cls_b, out2, BB*NQ_, NCLS_, DD, DD, NCLS_, 0);

  // -------- top-64 inverse-distance interpolation to parent points --------
  interp_kernel<<<BB*NP_, 256, 0, st>>>(parent, s_coord, S.pm, out1);
}
